// DecoderRNN2_37692632989826
// MI455X (gfx1250) — matrix operand order repaired and checked by reading, not compiled, not run on hardware
//
#include <hip/hip_runtime.h>
#include <hip/hip_bf16.h>

// ---------------------------------------------------------------------------
// DecoderRNN forward for MI455X (gfx1250, wave32).
//   k_xgates : emb gather + x@W_ih^T + b_ih            (fp32 VALU, 2.1 GFLOP)
//   k_hgates : features@W_hh^T + b_hh                  (fp32 VALU, tiny)
//   k_lstm   : elementwise cell-state scan over T      (hidden is NOT fed back
//              through W_hh in the reference)          -> hs in bf16
//   k_wmma_gemm<false> : hid = hs@W1^T + b1            (v_wmma_f32_16x16x32_bf16)
//   k_wmma_gemm<true>  : logits = hid@W2^T + b2        (v_wmma_f32_16x16x32_bf16,
//                        permuted (B,T,V) fp32 store)
// Wave tile 64x64 (16 WMMA tiles), block tile 128x256, K-step 32.
// A tile staged with global_load_async_to_lds_b128 (ASYNCcnt path, no VGPR hop);
// B tile (fp32 weights) converted in-flight with v_cvt_pk_bf16_f32.
// Workspace: ~22.3 MB.
// ---------------------------------------------------------------------------

typedef __attribute__((ext_vector_type(16))) __bf16 v16bf;
typedef __attribute__((ext_vector_type(8)))  __bf16 bf8v;
typedef __attribute__((ext_vector_type(8)))  float  f8v;
typedef __attribute__((ext_vector_type(8)))  float  v8f;

static constexpr int Bb = 32;      // batch
static constexpr int Tt = 64;      // seq len
static constexpr int Vv = 50257;   // vocab
static constexpr int Ee = 256;     // embed dim
static constexpr int Hh = 512;     // hidden dim

union AFrag { v16bf v; bf8v h[2]; };

// ---------------------------------------------------------------------------
// x_gates[t,b,g] = sum_e emb[cap[b, t==0?0:t-1], e] * W_ih[g,e] + b_ih[g]
// ---------------------------------------------------------------------------
__global__ __launch_bounds__(256) void k_xgates(const int* __restrict__ cap,
                                                const float* __restrict__ emb,
                                                const float* __restrict__ Wih,
                                                const float* __restrict__ bih,
                                                float* __restrict__ xg) {
    __shared__ float sx[Ee];
    const int r = blockIdx.x;            // r = t*32 + b
    const int t = r >> 5, b = r & 31;
    const int tsrc = (t == 0) ? 0 : (t - 1);
    const int tok = cap[b * Tt + tsrc];
    sx[threadIdx.x] = emb[(size_t)tok * Ee + threadIdx.x];
    __syncthreads();
    float acc[8] = {0.f, 0.f, 0.f, 0.f, 0.f, 0.f, 0.f, 0.f};
    for (int e = 0; e < Ee; e += 4) {
        const float4 xv = *(const float4*)&sx[e];
#pragma unroll
        for (int j = 0; j < 8; ++j) {
            const int g = threadIdx.x + j * 256;
            const float4 w = *(const float4*)&Wih[(size_t)g * Ee + e];
            acc[j] += xv.x * w.x + xv.y * w.y + xv.z * w.z + xv.w * w.w;
        }
    }
#pragma unroll
    for (int j = 0; j < 8; ++j) {
        const int g = threadIdx.x + j * 256;
        xg[(size_t)r * (4 * Hh) + g] = acc[j] + bih[g];
    }
}

// ---------------------------------------------------------------------------
// h_gates[b,g] = sum_h features[b,h] * W_hh[g,h] + b_hh[g]
// ---------------------------------------------------------------------------
__global__ __launch_bounds__(256) void k_hgates(const float* __restrict__ feat,
                                                const float* __restrict__ Whh,
                                                const float* __restrict__ bhh,
                                                float* __restrict__ hg) {
    __shared__ float sx[Hh];
    const int b = blockIdx.x;
    sx[threadIdx.x]       = feat[b * Hh + threadIdx.x];
    sx[threadIdx.x + 256] = feat[b * Hh + threadIdx.x + 256];
    __syncthreads();
    float acc[8] = {0.f, 0.f, 0.f, 0.f, 0.f, 0.f, 0.f, 0.f};
    for (int e = 0; e < Hh; e += 4) {
        const float4 xv = *(const float4*)&sx[e];
#pragma unroll
        for (int j = 0; j < 8; ++j) {
            const int g = threadIdx.x + j * 256;
            const float4 w = *(const float4*)&Whh[(size_t)g * Hh + e];
            acc[j] += xv.x * w.x + xv.y * w.y + xv.z * w.z + xv.w * w.w;
        }
    }
#pragma unroll
    for (int j = 0; j < 8; ++j) {
        const int g = threadIdx.x + j * 256;
        hg[(size_t)b * (4 * Hh) + g] = acc[j] + bhh[g];
    }
}

// ---------------------------------------------------------------------------
// LSTM cell-state scan: only c is carried; h_gates constant in t.
// ---------------------------------------------------------------------------
__global__ __launch_bounds__(256) void k_lstm(const float* __restrict__ xg,
                                              const float* __restrict__ hg,
                                              __bf16* __restrict__ hs) {
    const int tid = blockIdx.x * blockDim.x + threadIdx.x;  // 0..16383
    const int b = tid >> 9;
    const int h = tid & (Hh - 1);
    const float* hgr = hg + (size_t)b * (4 * Hh);
    const float hgi = hgr[h], hgf = hgr[Hh + h], hgg = hgr[2 * Hh + h], hgo = hgr[3 * Hh + h];
    float c = 0.f;
    for (int t = 0; t < Tt; ++t) {
        const float* row = xg + (size_t)(t * Bb + b) * (4 * Hh);
        const float gi = row[h] + hgi;
        const float gf = row[Hh + h] + hgf;
        const float gg = row[2 * Hh + h] + hgg;
        const float go = row[3 * Hh + h] + hgo;
        const float si = 1.f / (1.f + __expf(-gi));
        const float sf = 1.f / (1.f + __expf(-gf));
        const float so = 1.f / (1.f + __expf(-go));
        c = sf * c + si * tanhf(gg);
        hs[(size_t)(t * Bb + b) * Hh + h] = (__bf16)(so * tanhf(c));
    }
}

// ---------------------------------------------------------------------------
// bf16 WMMA GEMM: Out[m,n] = sum_k A[m,k] * Bw[n,k] + bias[n]
//   A  : M x K bf16 row-major  (staged via global_load_async_to_lds_b128)
//   Bw : N x K fp32 row-major  (converted to bf16 into LDS; OOB rows clamped —
//        their products only touch never-stored columns gn >= N)
// Block tile 128(M) x 256(N), 8 waves as 2x4, wave tile 64x64, K-step 32.
// Fragment layouts per CDNA5 ISA 7.12.2 (16-bit A 16x32; B 32x16; f32 C/D).
// ---------------------------------------------------------------------------
template <bool PERMUTE>
__global__ __launch_bounds__(256) void k_wmma_gemm(const __bf16* __restrict__ A,
                                                   const float* __restrict__ Bw,
                                                   const float* __restrict__ bias,
                                                   void* __restrict__ Out,
                                                   int M, int N, int K) {
    constexpr int LP = 40;                   // LDS row pitch (bf16), bank-spread
    __shared__ __bf16 sA[128 * LP];          // 10 KB
    __shared__ __bf16 sB[256 * LP];          // 20 KB

    const int tid  = threadIdx.x;
    const int bm   = blockIdx.x * 128;
    const int bn   = blockIdx.y * 256;
    const int wave = tid >> 5;
    const int lane = tid & 31;
    const int hf   = lane >> 4;              // lane half (0/1)
    const int ln   = lane & 15;
    const int wm   = (wave >> 2) * 64;       // wave M origin in block tile
    const int wn   = (wave & 3) * 64;        // wave N origin in block tile

    const v8f vz = {};
    v8f acc[4][4];
#pragma unroll
    for (int i = 0; i < 4; ++i)
#pragma unroll
        for (int j = 0; j < 4; ++j) acc[i][j] = vz;

    // --- A staging: 2 threads per row, async direct-to-LDS (no VGPR hop) ---
    const int arow = tid >> 1;
    const int akh  = (tid & 1) * 16;
    // Low 32 bits of a generic LDS pointer are the LDS byte offset
    // (flat-aperture rule).
    const unsigned ldsA = (unsigned)(uintptr_t)(&sA[arow * LP + akh]);
    const __bf16*  gA   = A + (size_t)(bm + arow) * K + akh;

    // --- B staging: 1 thread per row, clamped (OOB products never stored) ---
    const int brow = (bn + tid < N) ? (bn + tid) : (N - 1);
    const float* gB = Bw + (size_t)brow * K;

    for (int k0 = 0; k0 < K; k0 += 32) {
        // stage A tile (128 x 32 bf16): ASYNCcnt path, LDS written directly
        asm volatile(
            "global_load_async_to_lds_b128 %0, %1, off\n\t"
            "global_load_async_to_lds_b128 %0, %1, off offset:16"
            :
            : "v"(ldsA), "v"(gA + k0)
            : "memory");

        // stage B tile (256 x 32): fp32 -> bf16 on the fly
        {
            const f8v* src = (const f8v*)(gB + k0);
            const bf8v lo = __builtin_convertvector(src[0], bf8v);
            const bf8v hi = __builtin_convertvector(src[1], bf8v);
            const bf8v lo2 = __builtin_convertvector(src[2], bf8v);
            const bf8v hi2 = __builtin_convertvector(src[3], bf8v);
            *(bf8v*)&sB[tid * LP]     = lo;
            *(bf8v*)&sB[tid * LP + 8] = hi;
            *(bf8v*)&sB[tid * LP + 16] = lo2;
            *(bf8v*)&sB[tid * LP + 24] = hi2;
        }

        asm volatile("s_wait_asynccnt 0x0" ::: "memory");
        __syncthreads();

        // ---- load fragments (ISA 7.12.2 layouts) ----
        AFrag af[4], bfv[4];
#pragma unroll
        for (int mt = 0; mt < 4; ++mt) {
            const int rr = (wm + mt * 16 + ln) * LP;
            af[mt].h[0] = *(const bf8v*)&sA[rr + hf * 8];        // K = hf*8 + 0..7
            af[mt].h[1] = *(const bf8v*)&sA[rr + 16 + hf * 8];   // K = 16 + hf*8 + 0..7
        }
#pragma unroll
        for (int nt = 0; nt < 4; ++nt) {
            const int rr = (wn + nt * 16 + ln) * LP;
            bfv[nt].h[0] = *(const bf8v*)&sB[rr + hf * 8];       // K = hf*8 + 0..7
            bfv[nt].h[1] = *(const bf8v*)&sB[rr + 16 + hf * 8];  // K = 16 + hf*8 + 0..7
        }

#pragma unroll
        for (int mt = 0; mt < 4; ++mt)
#pragma unroll
            for (int nt = 0; nt < 4; ++nt)
                acc[mt][nt] = __builtin_amdgcn_wmma_f32_16x16x32_bf16(
                    false, af[mt].v, false, bfv[nt].v,
                    (short)0, acc[mt][nt], false, false);
        __syncthreads();
    }

    // ---- epilogue: bias + store (C/D layout: VGPR i -> M = hf*8 + i) ----
#pragma unroll
    for (int mt = 0; mt < 4; ++mt) {
#pragma unroll
        for (int nt = 0; nt < 4; ++nt) {
            const int gn = bn + wn + nt * 16 + ln;
            if (gn >= N) continue;
            const float bv = bias[gn];
#pragma unroll
            for (int i = 0; i < 8; ++i) {
                const int gm = bm + wm + mt * 16 + hf * 8 + i;
                const float v = acc[mt][nt][i] + bv;
                if constexpr (PERMUTE) {
                    const int t = gm >> 5, b = gm & 31;    // gm = t*32 + b
                    ((float*)Out)[((size_t)b * Tt + t) * (size_t)N + gn] = v;
                } else {
                    ((__bf16*)Out)[(size_t)gm * N + gn] = (__bf16)v;
                }
            }
        }
    }
}

// ---------------------------------------------------------------------------
extern "C" void kernel_launch(void* const* d_in, const int* in_sizes, int n_in,
                              void* d_out, int out_size, void* d_ws, size_t ws_size,
                              hipStream_t stream) {
    (void)in_sizes; (void)n_in; (void)out_size; (void)ws_size;

    const float* features = (const float*)d_in[0];   // (32,512)
    const int*   captions = (const int*)  d_in[1];   // (32,64)
    const float* emb      = (const float*)d_in[2];   // (50257,256)
    const float* W_ih     = (const float*)d_in[3];   // (2048,256)
    const float* W_hh     = (const float*)d_in[4];   // (2048,512)
    const float* b_ih     = (const float*)d_in[5];   // (2048,)
    const float* b_hh     = (const float*)d_in[6];   // (2048,)
    const float* W1       = (const float*)d_in[7];   // (1024,512)
    const float* b1       = (const float*)d_in[8];   // (1024,)
    const float* W2       = (const float*)d_in[9];   // (50257,1024)
    const float* b2       = (const float*)d_in[10];  // (50257,)
    float* out = (float*)d_out;                      // (32,64,50257)

    // workspace carve-up (all offsets 256B-aligned): ~22.3 MB total
    char* ws = (char*)d_ws;
    float*  xg  = (float*)(ws);                                          // 16 MB
    float*  hg  = (float*)(ws + (16u << 20));                            // 256 KB
    __bf16* hs  = (__bf16*)(ws + (16u << 20) + (1u << 18));              // 2 MB
    __bf16* hid = (__bf16*)(ws + (16u << 20) + (1u << 18) + (1u << 21)); // 4 MB

    k_xgates<<<dim3(Tt * Bb), dim3(256), 0, stream>>>(captions, emb, W_ih, b_ih, xg);
    k_hgates<<<dim3(Bb), dim3(256), 0, stream>>>(features, W_hh, b_hh, hg);
    k_lstm<<<dim3((Bb * Hh) / 256), dim3(256), 0, stream>>>(xg, hg, hs);

    // hid = hs @ W1^T + b1   : M=2048, N=1024, K=512
    k_wmma_gemm<false><<<dim3(2048 / 128, 1024 / 256), dim3(256), 0, stream>>>(
        hs, W1, b1, (void*)hid, 2048, 1024, 512);

    // logits = hid @ W2^T + b2 : M=2048, N=50257, K=1024 (permuted (B,T,V) store)
    k_wmma_gemm<true><<<dim3(2048 / 128, (Vv + 255) / 256), dim3(256), 0, stream>>>(
        hid, W2, b2, (void*)out, 2048, Vv, 1024);
}
